// TileRefineCostLookup_60112362274871
// MI455X (gfx1250) — matrix operand order repaired and checked by reading, not compiled, not run on hardware
//
#include <hip/hip_runtime.h>

// ---------------------------------------------------------------------------
// TileRefineCostLookup for MI455X (gfx1250): warp + grouped corr -> 3x(3x3
// conv + GroupNorm + SiLU) -> 1x1 heads.  Convs are implicit GEMM on
// v_wmma_f32_16x16x32_bf16 (bf16 A/B, f32 accumulate), activations staged as
// bf16 in workspace.  LDS tiles laid out so every WMMA fragment is two
// contiguous ds_load_b128s; next-K-step rows prefetched via global_prefetch.
// ---------------------------------------------------------------------------

#define DEVINL __device__ __forceinline__

constexpr int B_   = 2;
constexpr int C_   = 64;
constexpr int H_   = 320;
constexpr int W_   = 640;
constexpr int TF_  = 16;
constexpr int HID_ = 48;
constexpr int G_   = 8;
constexpr int D_   = 5;
constexpr int INCH_ = 2 * C_ + TF_ + 4 + G_ * D_;   // 188
constexpr int HW_   = H_ * W_;                      // 204800
constexpr int NPIX_ = B_ * HW_;                     // 409600
static_assert(INCH_ == 188, "in-channel count");

typedef __attribute__((ext_vector_type(16))) __bf16 v16bf;
typedef __attribute__((ext_vector_type(8)))  float  v8f;

union BfFrag { v16bf v; unsigned short s[16]; };

DEVINL unsigned short f32_to_bf16(float f) {
  unsigned int u = __float_as_uint(f);
  u += 0x7FFFu + ((u >> 16) & 1u);          // round-to-nearest-even
  return (unsigned short)(u >> 16);
}
DEVINL float bf16_to_f32(unsigned short h) {
  return __uint_as_float(((unsigned int)h) << 16);
}
DEVINL float sigmoid_(float x) { return 1.0f / (1.0f + expf(-x)); }
DEVINL float softplus_(float x) {
  if (x > 20.0f) return x;
  if (x < -20.0f) return expf(x);
  return log1pf(expf(x));
}

// ---------------------------------------------------------------------------
// Kernel 1: warp fR, grouped correlation, assemble 188-ch bf16 input tensor.
// X layout: [B][188][H][W] bf16; channels = fL(64), fR_w(64), feat(16),
// d, sx, sy, conf, cost(40 = g*5 + off).
// ---------------------------------------------------------------------------
__global__ __launch_bounds__(256) void prep_kernel(
    const float* __restrict__ d, const float* __restrict__ sx,
    const float* __restrict__ sy, const float* __restrict__ feat,
    const float* __restrict__ conf, const float* __restrict__ fL,
    const float* __restrict__ fR, unsigned short* __restrict__ X) {
  int idx = blockIdx.x * 256 + threadIdx.x;
  if (idx >= NPIX_) return;
  int w = idx % W_;
  int h = (idx / W_) % H_;
  int b = idx / HW_;
  int hw = h * W_ + w;
  int ib = b * HW_ + hw;

  float dv = d[ib];

#define XS(ch, val) X[((size_t)b * INCH_ + (ch)) * HW_ + hw] = f32_to_bf16(val)

  const float* fLrow = fL + (size_t)b * C_ * HW_ + (size_t)h * W_;
  const float* fRrow = fR + (size_t)b * C_ * HW_ + (size_t)h * W_;

  for (int c = 0; c < C_; ++c) XS(c, fLrow[(size_t)c * HW_ + w]);
  for (int c = 0; c < TF_; ++c) XS(2 * C_ + c, feat[((size_t)b * TF_ + c) * HW_ + hw]);
  XS(144, dv);
  XS(145, sx[ib]);
  XS(146, sy[ib]);
  XS(147, conf[ib]);

  for (int i = 0; i < D_; ++i) {
    float off = (float)(i - 2);
    float xs = (float)w - (dv + off);
    xs = fminf(fmaxf(xs, 0.0f), (float)(W_ - 1));
    float x0 = floorf(xs);
    float fr = xs - x0;
    int i0 = (int)x0;
    int i1 = min(i0 + 1, W_ - 1);
    float accg[G_];
#pragma unroll
    for (int g = 0; g < G_; ++g) accg[g] = 0.0f;
    for (int c = 0; c < C_; ++c) {
      float g0 = fRrow[(size_t)c * HW_ + i0];
      float g1 = fRrow[(size_t)c * HW_ + i1];
      float v = g0 * (1.0f - fr) + g1 * fr;
      if (i == 2) XS(C_ + c, v);                       // fR_w (offset 0)
      accg[c >> 3] += fLrow[(size_t)c * HW_ + w] * v;  // cg = 8 per group
    }
#pragma unroll
    for (int g = 0; g < G_; ++g) XS(148 + g * D_ + i, accg[g] * 0.125f);
  }
#undef XS
}

// ---------------------------------------------------------------------------
// Kernel 2: 3x3 conv (pad 1) as implicit GEMM on WMMA bf16.
//   out[b][m][h][w] = sum_{ic,dy,dx} W[m][ic][dy][dx] * X[b][ic][h+dy-1][w+dx-1]
// Block = 256 threads = 8 waves; tile = 48 outch x 128 pixels of one row.
// Wave w owns 16 pixels (N) and loops 3 M-tiles; K staged 64 at a time.
// LDS layout (padded stride 72 half-words = 144 B, 16 B aligned, bank-spread):
//   As[m][k]  m-major  -> A frag = 1 contiguous 32 B run per lane
//   Bs[n][k]  n-major  -> B frag = 1 contiguous 32 B run per lane
// ---------------------------------------------------------------------------
template <int CIN>
__global__ __launch_bounds__(256) void conv3x3_wmma_kernel(
    const unsigned short* __restrict__ X,  // [B][CIN][H][W] bf16
    const float* __restrict__ Wt,          // [48][CIN][3][3] f32
    float* __restrict__ Y) {               // [B][48][H][W] f32
  constexpr int KTOT = CIN * 9;
  constexpr int KSTEPS = (KTOT + 63) / 64;
  constexpr int WT = W_ / 128;             // 5 pixel-tiles per row
  constexpr int LDA = 72;                  // padded K-stride (half-words)
  constexpr int LDB = 72;

  __shared__ unsigned short As[HID_ * LDA];   // 6.9 KB
  __shared__ unsigned short Bs[128 * LDB];    // 18.4 KB

  int tid = threadIdx.x;
  int wave = tid >> 5;
  int lane = tid & 31;

  int blk = blockIdx.x;
  int w0 = (blk % WT) * 128;
  int h  = (blk / WT) % H_;
  int b  = blk / (WT * H_);

  v8f acc[3] = {};

  for (int ks = 0; ks < KSTEPS; ++ks) {
    int k0 = ks * 64;
    // --- stage A: 48x64 bf16 weights (zero-pad past KTOT) ---
    for (int i = tid; i < HID_ * 64; i += 256) {
      int m = i >> 6;
      int kk = i & 63;
      int k = k0 + kk;
      float wv = 0.0f;
      if (k < KTOT) {
        int ic = k / 9, r = k % 9;
        wv = Wt[((size_t)(m * CIN + ic) * 3 + (r / 3)) * 3 + (r % 3)];
      }
      As[m * LDA + kk] = f32_to_bf16(wv);
    }
    // --- stage B: 64x128 im2col patch; global reads n-fast (coalesced),
    //     LDS writes n-major ---
    for (int i = tid; i < 64 * 128; i += 256) {
      int kk = i >> 7;          // 0..63
      int n  = i & 127;
      int k = k0 + kk;
      unsigned short v = 0;
      if (k < KTOT) {
        int ic = k / 9, r = k % 9;
        int hy = h + (r / 3) - 1;
        int wx = w0 + n + (r % 3) - 1;
        if (hy >= 0 && hy < H_ && wx >= 0 && wx < W_)
          v = X[((size_t)(b * CIN + ic) * H_ + hy) * W_ + wx];
      }
      Bs[n * LDB + kk] = v;
    }
    // --- prefetch next K-step's activation rows (global_prefetch path) ---
    if (ks + 1 < KSTEPS) {
      int k = k0 + 64 + (tid >> 2);       // 256 threads cover 64 k-values x 4
      if (k < KTOT) {
        int ic = k / 9, r = k % 9;
        int hy = h + (r / 3) - 1;
        int wx = w0 + ((tid & 3) << 5) + (r % 3) - 1;
        if (hy >= 0 && hy < H_ && wx >= 0 && wx < W_)
          __builtin_prefetch(&X[((size_t)(b * CIN + ic) * H_ + hy) * W_ + wx]);
      }
    }
    __syncthreads();

    int nn = (wave << 4) + (lane & 15);   // this wave's pixel column
    int half = lane >> 4;                 // lane half selects K sub-range
#pragma unroll
    for (int s = 0; s < 2; ++s) {
      int kbase = s * 32;
      // B fragment: k = kbase + half*16 + e  (contiguous 32 B)
      BfFrag bfr;
#pragma unroll
      for (int e = 0; e < 16; ++e)
        bfr.s[e] = Bs[nn * LDB + kbase + half * 16 + e];
      // A fragments: k = kbase + half*16 + e  (contiguous 32 B)
#pragma unroll
      for (int mi = 0; mi < 3; ++mi) {
        BfFrag afr;
        int m = mi * 16 + (lane & 15);
#pragma unroll
        for (int e = 0; e < 16; ++e) {
          int k = kbase + half * 16 + e;
          afr.s[e] = As[m * LDA + k];
        }
        acc[mi] = __builtin_amdgcn_wmma_f32_16x16x32_bf16(
            false, afr.v, false, bfr.v, (short)0, acc[mi], false, false);
      }
    }
    __syncthreads();
  }

  // D layout: reg r holds M = r + 8*(lane>=16), N = lane%16
  int n = w0 + (wave << 4) + (lane & 15);
  int mh = (lane >> 4) << 3;
#pragma unroll
  for (int mi = 0; mi < 3; ++mi)
#pragma unroll
    for (int r = 0; r < 8; ++r) {
      int m = mi * 16 + r + mh;
      Y[((size_t)(b * HID_ + m) * H_ + h) * W_ + n] = acc[mi][r];
    }
}

// ---------------------------------------------------------------------------
// GroupNorm support: zero stats, block-partial sums + atomics, then
// normalize + SiLU + bf16 re-emit.
// ---------------------------------------------------------------------------
__global__ void init_stats_kernel(float* __restrict__ stats) {
  if (threadIdx.x < 32) stats[threadIdx.x] = 0.0f;
}

__global__ __launch_bounds__(256) void gn_stats_kernel(
    const float* __restrict__ Y, float* __restrict__ stats) {
  constexpr int CHUNK = HW_ / 16;                // 12800
  int plane = blockIdx.x >> 4;                   // b*48 + c
  int chunk = blockIdx.x & 15;
  int b = plane / HID_, c = plane % HID_;
  int g = b * 8 + c / 6;
  const float* base = Y + (size_t)plane * HW_ + (size_t)chunk * CHUNK;
  float s = 0.0f, s2 = 0.0f;
  for (int i = threadIdx.x; i < CHUNK; i += 256) {
    float v = base[i];
    s += v;
    s2 += v * v;
  }
  __shared__ float rs[256], rs2[256];
  rs[threadIdx.x] = s;
  rs2[threadIdx.x] = s2;
  __syncthreads();
  for (int off = 128; off > 0; off >>= 1) {
    if (threadIdx.x < off) {
      rs[threadIdx.x] += rs[threadIdx.x + off];
      rs2[threadIdx.x] += rs2[threadIdx.x + off];
    }
    __syncthreads();
  }
  if (threadIdx.x == 0) {
    atomicAdd(&stats[g * 2 + 0], rs[0]);
    atomicAdd(&stats[g * 2 + 1], rs2[0]);
  }
}

__global__ __launch_bounds__(256) void gn_act_kernel(
    const float* __restrict__ Y, const float* __restrict__ stats,
    const float* __restrict__ gamma, const float* __restrict__ beta,
    unsigned short* __restrict__ Xo) {
  int idx = blockIdx.x * 256 + threadIdx.x;      // B*48*HW = 19,660,800 exact
  int c = (idx / HW_) % HID_;
  int b = idx / (HID_ * HW_);
  int g = b * 8 + c / 6;
  const float cnt = 6.0f * (float)HW_;
  float m = stats[g * 2 + 0] / cnt;
  float var = stats[g * 2 + 1] / cnt - m * m;
  float inv = rsqrtf(var + 1e-5f);
  float v = (Y[idx] - m) * inv * gamma[c] + beta[c];
  Xo[idx] = f32_to_bf16(v * sigmoid_(v));        // SiLU
}

// ---------------------------------------------------------------------------
// Kernel 5: 1x1 heads + output math.  Weights (20x48) staged in LDS.
// Output tuple flat: d(409600) sx(409600) sy(409600) feat(6553600) conf(409600)
// ---------------------------------------------------------------------------
__global__ __launch_bounds__(256) void head_kernel(
    const unsigned short* __restrict__ Xh, const float* __restrict__ d,
    const float* __restrict__ sx, const float* __restrict__ sy,
    const float* __restrict__ feat, const float* __restrict__ conf,
    const float* __restrict__ hd_w, const float* __restrict__ hd_b,
    const float* __restrict__ hsx_w, const float* __restrict__ hsx_b,
    const float* __restrict__ hsy_w, const float* __restrict__ hsy_b,
    const float* __restrict__ hc_w, const float* __restrict__ hc_b,
    const float* __restrict__ hf_w, const float* __restrict__ hf_b,
    float* __restrict__ out) {
  __shared__ float wbuf[20 * HID_];
  __shared__ float bbuf[20];
  int tid = threadIdx.x;
  for (int i = tid; i < HID_; i += 256) {
    wbuf[0 * HID_ + i] = hd_w[i];
    wbuf[1 * HID_ + i] = hsx_w[i];
    wbuf[2 * HID_ + i] = hsy_w[i];
    wbuf[3 * HID_ + i] = hc_w[i];
  }
  for (int i = tid; i < 16 * HID_; i += 256) wbuf[4 * HID_ + i] = hf_w[i];
  if (tid == 0) {
    bbuf[0] = hd_b[0]; bbuf[1] = hsx_b[0]; bbuf[2] = hsy_b[0]; bbuf[3] = hc_b[0];
  }
  if (tid < 16) bbuf[4 + tid] = hf_b[tid];
  __syncthreads();

  int idx = blockIdx.x * 256 + tid;
  if (idx >= NPIX_) return;
  int hw = idx % HW_;
  int b = idx / HW_;

  float hv[HID_];
#pragma unroll
  for (int c = 0; c < HID_; ++c)
    hv[c] = bf16_to_f32(Xh[((size_t)b * HID_ + c) * HW_ + hw]);

  float acc[20];
#pragma unroll
  for (int r = 0; r < 20; ++r) {
    float s = bbuf[r];
#pragma unroll
    for (int c = 0; c < HID_; ++c) s += wbuf[r * HID_ + c] * hv[c];
    acc[r] = s;
  }

  int p = b * HW_ + hw;
  float dv = d[p], sxv = sx[p], syv = sy[p], cv = conf[p];
  out[p]               = softplus_(acc[0] + dv);
  out[NPIX_ + p]       = sxv + 0.1f * acc[1];
  out[2 * NPIX_ + p]   = syv + 0.1f * acc[2];
  out[19 * NPIX_ + p]  = sigmoid_(acc[3] + 2.0f * cv - 1.0f);   // after 16-ch feat
#pragma unroll
  for (int j = 0; j < TF_; ++j)
    out[3 * NPIX_ + ((size_t)b * TF_ + j) * HW_ + hw] =
        feat[((size_t)b * TF_ + j) * HW_ + hw] + acc[4 + j];
}

// ---------------------------------------------------------------------------
extern "C" void kernel_launch(void* const* d_in, const int* in_sizes, int n_in,
                              void* d_out, int out_size, void* d_ws, size_t ws_size,
                              hipStream_t stream) {
  const float* d    = (const float*)d_in[0];
  const float* sx   = (const float*)d_in[1];
  const float* sy   = (const float*)d_in[2];
  const float* feat = (const float*)d_in[3];
  const float* conf = (const float*)d_in[4];
  const float* fL   = (const float*)d_in[5];
  const float* fR   = (const float*)d_in[6];
  const float* w1   = (const float*)d_in[7];
  const float* w2   = (const float*)d_in[8];
  const float* w3   = (const float*)d_in[9];
  const float* gng[3] = {(const float*)d_in[10], (const float*)d_in[12], (const float*)d_in[14]};
  const float* gnb[3] = {(const float*)d_in[11], (const float*)d_in[13], (const float*)d_in[15]};
  const float* hd_w  = (const float*)d_in[16];
  const float* hd_b  = (const float*)d_in[17];
  const float* hsx_w = (const float*)d_in[18];
  const float* hsx_b = (const float*)d_in[19];
  const float* hsy_w = (const float*)d_in[20];
  const float* hsy_b = (const float*)d_in[21];
  const float* hc_w  = (const float*)d_in[22];
  const float* hc_b  = (const float*)d_in[23];
  const float* hf_w  = (const float*)d_in[24];
  const float* hf_b  = (const float*)d_in[25];
  float* out = (float*)d_out;

  // workspace carve-up
  char* ws = (char*)d_ws;
  size_t off = 0;
  unsigned short* X1 = (unsigned short*)(ws + off);
  off += (size_t)B_ * INCH_ * HW_ * sizeof(unsigned short);   // 154.0 MB
  float* Y = (float*)(ws + off);
  off += (size_t)B_ * HID_ * HW_ * sizeof(float);             //  78.6 MB
  unsigned short* Xh = (unsigned short*)(ws + off);
  off += (size_t)B_ * HID_ * HW_ * sizeof(unsigned short);    //  39.3 MB
  float* stats = (float*)(ws + off);                          // 32 floats

  const int conv_grid = B_ * H_ * (W_ / 128);                 // 3200
  const int pix_grid  = (NPIX_ + 255) / 256;                  // 1600
  const int act_grid  = (B_ * HID_ * HW_) / 256;              // 76800
  const int st_grid   = B_ * HID_ * 16;                       // 1536

  prep_kernel<<<pix_grid, 256, 0, stream>>>(d, sx, sy, feat, conf, fL, fR, X1);

  // layer 1: 188 -> 48
  conv3x3_wmma_kernel<INCH_><<<conv_grid, 256, 0, stream>>>(X1, w1, Y);
  init_stats_kernel<<<1, 32, 0, stream>>>(stats);
  gn_stats_kernel<<<st_grid, 256, 0, stream>>>(Y, stats);
  gn_act_kernel<<<act_grid, 256, 0, stream>>>(Y, stats, gng[0], gnb[0], Xh);

  // layer 2: 48 -> 48
  conv3x3_wmma_kernel<HID_><<<conv_grid, 256, 0, stream>>>(Xh, w2, Y);
  init_stats_kernel<<<1, 32, 0, stream>>>(stats);
  gn_stats_kernel<<<st_grid, 256, 0, stream>>>(Y, stats);
  gn_act_kernel<<<act_grid, 256, 0, stream>>>(Y, stats, gng[1], gnb[1], Xh);

  // layer 3: 48 -> 48
  conv3x3_wmma_kernel<HID_><<<conv_grid, 256, 0, stream>>>(Xh, w3, Y);
  init_stats_kernel<<<1, 32, 0, stream>>>(stats);
  gn_stats_kernel<<<st_grid, 256, 0, stream>>>(Y, stats);
  gn_act_kernel<<<act_grid, 256, 0, stream>>>(Y, stats, gng[2], gnb[2], Xh);

  // heads + output math
  head_kernel<<<pix_grid, 256, 0, stream>>>(Xh, d, sx, sy, feat, conf,
                                            hd_w, hd_b, hsx_w, hsx_b,
                                            hsy_w, hsy_b, hc_w, hc_b,
                                            hf_w, hf_b, out);
}
